// AttentionBlock_41729902248378
// MI455X (gfx1250) — matrix operand order repaired and checked by reading, not compiled, not run on hardware
//
#include <hip/hip_runtime.h>
#include <hip/hip_fp16.h>

// ---------------------------------------------------------------------------
// Problem constants (from reference): B=4, C=256, H=W=64, 32 groups, 4 heads
// ---------------------------------------------------------------------------
constexpr int BSZ = 4;
constexpr int C   = 256;
constexpr int L   = 4096;        // H*W
constexpr int G   = 32;
constexpr int CPG = C / G;       // 8
constexpr int NH  = 4;
constexpr int D   = 64;          // head dim
constexpr float EPS = 1e-5f;

typedef _Float16 v16h __attribute__((ext_vector_type(16)));
typedef _Float16 v8h  __attribute__((ext_vector_type(8)));
typedef float    v8f  __attribute__((ext_vector_type(8)));
typedef unsigned int u32x4 __attribute__((ext_vector_type(4)));
typedef int          i32x4 __attribute__((ext_vector_type(4)));
typedef int          i32x8 __attribute__((ext_vector_type(8)));

// ---------------------------------------------------------------------------
// WMMA helper: D(16x16,f32) = A(16x32,f16) * B(32x16,f16) + C
// ---------------------------------------------------------------------------
__device__ __forceinline__ v8f wmma_f16(v16h a, v16h b, v8f c) {
  return __builtin_amdgcn_wmma_f32_16x16x32_f16(false, a, false, b, (short)0, c,
                                                false, false);
}

// A fragment (16x32 f16) from a row-major source, leading dim lda.
// Lanes 0-15 hold row M=lane, K=0..15;
// lanes 16-31 hold row M=lane-16, K=16..31 (contiguous, matches B fragments).
__device__ __forceinline__ v16h load_A_rowmajor(const _Float16* A, int lda) {
  const int lane = threadIdx.x & 31;
  const _Float16* r = A + (lane & 15) * lda + ((lane >> 4) << 4);
  v8h lo = *(const v8h*)(r);
  v8h hi = *(const v8h*)(r + 8);
  return __builtin_shufflevector(lo, hi, 0, 1, 2, 3, 4, 5, 6, 7,
                                 8, 9, 10, 11, 12, 13, 14, 15);
}

// B fragment (32x16 f16) from an "N-major" tile: tile[n][k], row width 32.
// Lane holds col N (=tile row), K 0..15 (lanes 0-15) / 16..31 (lanes 16-31):
// one contiguous 32B vector load per lane.
__device__ __forceinline__ v16h load_B_nmajor(const _Float16* T, int ncol) {
  const int lane = threadIdx.x & 31;
  return *(const v16h*)(T + (ncol + (lane & 15)) * 32 + ((lane >> 4) << 4));
}

// ---------------------------------------------------------------------------
// Tensor Data Mover: async 2-D f16 tile load (global -> LDS), tracked with
// TENSORcnt. D# built per CDNA5 ISA ch.8: group0 = {flags, lds_addr,
// global_addr, type=2}; group1 = {data_size=2B, tensor dims, tile dims,
// dim0 stride}; groups 2/3 zero (2-D tile).
// ---------------------------------------------------------------------------
__device__ __forceinline__ void tdm_load_2d(const _Float16* lds_dst,
                                            const _Float16* gsrc,
                                            unsigned tile_w, unsigned tile_h,
                                            unsigned stride_elems) {
  const unsigned long long ga = (unsigned long long)(uintptr_t)gsrc;
  u32x4 g0;
  g0[0] = 1u;                                           // count=1 (valid)
  g0[1] = (unsigned)(uintptr_t)lds_dst;                 // LDS byte address
  g0[2] = (unsigned)(ga & 0xFFFFFFFFu);                 // global_addr[31:0]
  g0[3] = (unsigned)((ga >> 32) & 0x1FFFFFFu) | (2u << 30);  // [56:32] | type=2
  i32x8 g1;
  g1[0] = 1 << 16;                                      // data_size=2 bytes
  g1[1] = (int)((tile_w & 0xFFFFu) << 16);              // tensor_dim0 lo
  g1[2] = (int)((tile_w >> 16) | ((tile_h & 0xFFFFu) << 16));  // d0 hi | d1 lo
  g1[3] = (int)((tile_h >> 16) | (tile_w << 16));       // d1 hi | tile_dim0
  g1[4] = (int)tile_h;                                  // tile_dim1 (tile_dim2=0)
  g1[5] = (int)stride_elems;                            // tensor_dim0_stride lo
  g1[6] = 0;
  g1[7] = 0;
  i32x4 z4 = {};
#if defined(__clang_major__) && (__clang_major__ >= 23)
  i32x8 z8 = {};
  __builtin_amdgcn_tensor_load_to_lds(g0, g1, z4, z4, z8, 0);
#else
  __builtin_amdgcn_tensor_load_to_lds(g0, g1, z4, z4, 0);
#endif
}

// ---------------------------------------------------------------------------
// K0: convert weights f32 -> f16
// ---------------------------------------------------------------------------
__global__ void k_convert_w(const float* __restrict__ qkv_w,
                            const float* __restrict__ proj_w,
                            _Float16* __restrict__ wq,
                            _Float16* __restrict__ wp) {
  int i = blockIdx.x * 256 + threadIdx.x;
  if (i < 3 * C * C) wq[i] = (_Float16)qkv_w[i];
  if (i < C * C)     wp[i] = (_Float16)proj_w[i];
}

// ---------------------------------------------------------------------------
// K1: group-norm statistics: one block per (b, g), 32768 contiguous floats
// ---------------------------------------------------------------------------
__global__ void k_gn_stats(const float* __restrict__ x, float* __restrict__ stats) {
  const int bg = blockIdx.x;
  const float* p = x + (size_t)bg * CPG * L;
  float s = 0.f, ss = 0.f;
  for (int i = threadIdx.x; i < CPG * L; i += 256) {
    float v = p[i];
    s += v; ss += v * v;
  }
  __shared__ float sm[256], sm2[256];
  sm[threadIdx.x] = s; sm2[threadIdx.x] = ss;
  __syncthreads();
  for (int st = 128; st > 0; st >>= 1) {
    if (threadIdx.x < st) {
      sm[threadIdx.x]  += sm[threadIdx.x + st];
      sm2[threadIdx.x] += sm2[threadIdx.x + st];
    }
    __syncthreads();
  }
  if (threadIdx.x == 0) {
    const float inv_n = 1.0f / (float)(CPG * L);
    float mean = sm[0] * inv_n;
    float var  = sm2[0] * inv_n - mean * mean;
    stats[bg * 2 + 0] = mean;
    stats[bg * 2 + 1] = rsqrtf(var + EPS);
  }
}

// ---------------------------------------------------------------------------
// K2: apply GN and transpose: x (B,C,L) f32 -> ht (B,L,C) f16 via LDS tile.
// Block handles 16 l-rows x 256 channels.
// ---------------------------------------------------------------------------
__global__ void k_gn_apply_t(const float* __restrict__ x,
                             const float* __restrict__ stats,
                             const float* __restrict__ gw,
                             const float* __restrict__ gb,
                             _Float16* __restrict__ ht) {
  __shared__ _Float16 tile[16 * 256];
  const int l0 = blockIdx.x * 16;
  const int b  = blockIdx.y;
  const int tid = threadIdx.x;
#pragma unroll
  for (int k = 0; k < 16; ++k) {
    const int e = tid + k * 256;
    const int c = e >> 4;
    const int li = e & 15;
    const int bg = b * G + (c >> 3);
    const float mean = stats[bg * 2 + 0];
    const float rstd = stats[bg * 2 + 1];
    const float v = x[((size_t)b * C + c) * L + l0 + li];
    tile[li * 256 + c] = (_Float16)(((v - mean) * rstd) * gw[c] + gb[c]);
  }
  __syncthreads();
#pragma unroll
  for (int k = 0; k < 16; ++k) {
    const int e = tid + k * 256;
    const int li = e >> 8;
    const int c = e & 255;
    ht[((size_t)b * L + l0 + li) * C + c] = tile[li * 256 + c];
  }
}

// ---------------------------------------------------------------------------
// K3: QKV GEMM. qkv[o,l] = sum_c W[o,c]*ht[l,c] + bias[o].
// One wave per block -> 16(M=o) x 64(N=l) tile. B tiles (64 l-rows x 32 c)
// arrive via TDM double-buffered in LDS; fragments are contiguous reads.
// Scatter: Q,K head-major (L,D); V transposed head-major (D,L).
// ---------------------------------------------------------------------------
__global__ __launch_bounds__(32) void k_qkv_gemm(
    const _Float16* __restrict__ W, const _Float16* __restrict__ Ht,
    const float* __restrict__ bias, _Float16* __restrict__ Q,
    _Float16* __restrict__ Kx, _Float16* __restrict__ Vt) {
  __shared__ alignas(32) _Float16 Ts[2][64 * 32];
  const int o0 = blockIdx.x * 16;
  const int l0 = blockIdx.y * 64;
  const int b  = blockIdx.z;
  const int lane = threadIdx.x;
  const _Float16* Ap = W + o0 * C;
  const _Float16* Bp = Ht + ((size_t)b * L + l0) * C;   // +c for column offset
  v8f acc0 = {}, acc1 = {}, acc2 = {}, acc3 = {};

  tdm_load_2d(Ts[0], Bp, 32, 64, C);
  for (int kk = 0; kk < C; kk += 32) {
    const int cur = (kk >> 5) & 1;
    if (kk + 32 < C) {
      tdm_load_2d(Ts[cur ^ 1], Bp + kk + 32, 32, 64, C);
      __builtin_amdgcn_s_wait_tensorcnt(1);
    } else {
      __builtin_amdgcn_s_wait_tensorcnt(0);
    }
    __syncthreads();
    const v16h a = load_A_rowmajor(Ap + kk, C);
    const _Float16* T = Ts[cur];
    acc0 = wmma_f16(a, load_B_nmajor(T, 0),  acc0);
    acc1 = wmma_f16(a, load_B_nmajor(T, 16), acc1);
    acc2 = wmma_f16(a, load_B_nmajor(T, 32), acc2);
    acc3 = wmma_f16(a, load_B_nmajor(T, 48), acc3);
    __syncthreads();
  }
  const int n = lane & 15;
  const int mb = (lane >> 4) << 3;
  v8f accs[4] = {acc0, acc1, acc2, acc3};
#pragma unroll
  for (int j = 0; j < 4; ++j) {
#pragma unroll
    for (int r = 0; r < 8; ++r) {
      const int o = o0 + mb + r;
      const int l = l0 + j * 16 + n;
      const float val = accs[j][r] + bias[o];
      const int t = o >> 8;            // 0=q 1=k 2=v (16-row tile never crosses)
      const int c = o & (C - 1);
      const int hd = c >> 6, dd = c & 63;
      if (t == 2) {
        Vt[((size_t)(b * NH + hd) * D + dd) * L + l] = (_Float16)val;
      } else {
        _Float16* dst = (t == 0) ? Q : Kx;
        dst[((size_t)(b * NH + hd) * L + l) * D + dd] = (_Float16)val;
      }
    }
  }
}

// ---------------------------------------------------------------------------
// K4: flash attention. One wave = 16 query rows of one (b, head); sweeps all
// 4096 keys 32 at a time. K tiles (32 keys x 64 d from (L,D)) and V tiles
// (64 d x 32 keys from (D,L)) are TDM double-buffered; online softmax in f32.
// Output stored f16 transposed to (B, L, C) for the projection GEMM.
// ---------------------------------------------------------------------------
__global__ __launch_bounds__(32) void k_attn(const _Float16* __restrict__ Q,
                                             const _Float16* __restrict__ Km,
                                             const _Float16* __restrict__ Vt,
                                             _Float16* __restrict__ Ot) {
  __shared__ alignas(32) _Float16 Ks[2][32 * 64];   // [key][d]
  __shared__ alignas(32) _Float16 Vs[2][64 * 32];   // [d][key]
  __shared__ alignas(32) _Float16 Ps[16 * 32];      // [q][key]
  const int q0 = blockIdx.x * 16;
  const int bh = blockIdx.y;                        // b*NH + head
  const _Float16* Qp = Q  + (size_t)bh * L * D;
  const _Float16* Kp = Km + (size_t)bh * L * D;
  const _Float16* Vp = Vt + (size_t)bh * D * L;
  const int lane = threadIdx.x;
  const int n    = lane & 15;
  const int mb   = (lane >> 4) << 3;
  const int kb8  = (lane >> 4) << 3;
  const int kb16 = (lane >> 4) << 4;

  // Q A-fragments for d=0..63, pre-scaled by d^-1/2 (=2^-3, exact in f16).
  v16h qa0 = load_A_rowmajor(Qp + (size_t)q0 * D, D);
  v16h qa1 = load_A_rowmajor(Qp + (size_t)q0 * D + 32, D);
#pragma unroll
  for (int i = 0; i < 16; ++i) {
    qa0[i] *= (_Float16)0.125f;
    qa1[i] *= (_Float16)0.125f;
  }

  v8f oa0 = {}, oa1 = {}, oa2 = {}, oa3 = {};
  float mrow[8], lrow[8];
#pragma unroll
  for (int r = 0; r < 8; ++r) { mrow[r] = -3.0e38f; lrow[r] = 0.f; }

  tdm_load_2d(Ks[0], Kp, 64, 32, D);
  tdm_load_2d(Vs[0], Vp, 32, 64, L);
  for (int key0 = 0; key0 < L; key0 += 32) {
    const int cur = (key0 >> 5) & 1;
    if (key0 + 32 < L) {
      tdm_load_2d(Ks[cur ^ 1], Kp + (size_t)(key0 + 32) * D, 64, 32, D);
      tdm_load_2d(Vs[cur ^ 1], Vp + key0 + 32, 32, 64, L);
      __builtin_amdgcn_s_wait_tensorcnt(2);   // current K+V tiles complete
    } else {
      __builtin_amdgcn_s_wait_tensorcnt(0);
    }
    __syncthreads();

    // K^T B-fragments: contiguous 32B LDS reads from the [key][d] tile.
    const _Float16* Kt = Ks[cur];
    const v16h kb0lo = *(const v16h*)(Kt + n * 64 + kb16);
    const v16h kb0hi = *(const v16h*)(Kt + n * 64 + 32 + kb16);
    const v16h kb1lo = *(const v16h*)(Kt + (16 + n) * 64 + kb16);
    const v16h kb1hi = *(const v16h*)(Kt + (16 + n) * 64 + 32 + kb16);

    v8f s0 = {}, s1 = {};
    s0 = wmma_f16(qa0, kb0lo, s0);
    s0 = wmma_f16(qa1, kb0hi, s0);
    s1 = wmma_f16(qa0, kb1lo, s1);
    s1 = wmma_f16(qa1, kb1hi, s1);

    // Online softmax; row values live in one 16-lane half (wave32).
#pragma unroll
    for (int r = 0; r < 8; ++r) {
      float a = s0[r], bb = s1[r];
      float mx = fmaxf(a, bb);
#pragma unroll
      for (int msk = 8; msk >= 1; msk >>= 1) mx = fmaxf(mx, __shfl_xor(mx, msk, 32));
      const float mnew = fmaxf(mrow[r], mx);
      const float corr = __expf(mrow[r] - mnew);
      const float p0 = __expf(a - mnew);
      const float p1 = __expf(bb - mnew);
      float rs = p0 + p1;
#pragma unroll
      for (int msk = 8; msk >= 1; msk >>= 1) rs += __shfl_xor(rs, msk, 32);
      lrow[r] = lrow[r] * corr + rs;
      mrow[r] = mnew;
      oa0[r] *= corr; oa1[r] *= corr; oa2[r] *= corr; oa3[r] *= corr;
      Ps[(mb + r) * 32 + n]      = (_Float16)p0;
      Ps[(mb + r) * 32 + 16 + n] = (_Float16)p1;
    }
    __syncthreads();

    // O += P * V: P re-read as A-fragment (contiguous), V frags contiguous.
    const v16h pa = load_A_rowmajor(Ps, 32);
    const _Float16* Vc = Vs[cur];
    oa0 = wmma_f16(pa, load_B_nmajor(Vc, 0),  oa0);
    oa1 = wmma_f16(pa, load_B_nmajor(Vc, 16), oa1);
    oa2 = wmma_f16(pa, load_B_nmajor(Vc, 32), oa2);
    oa3 = wmma_f16(pa, load_B_nmajor(Vc, 48), oa3);
    __syncthreads();
  }

  // Finalize and store transposed into (B, L, C): channel = head*64 + d.
  const int b  = bh >> 2;
  const int hd = bh & 3;
#pragma unroll
  for (int r = 0; r < 8; ++r) {
    const float inv = 1.0f / lrow[r];
    _Float16* orow = Ot + ((size_t)b * L + q0 + mb + r) * C + hd * 64 + n;
    orow[0]  = (_Float16)(oa0[r] * inv);
    orow[16] = (_Float16)(oa1[r] * inv);
    orow[32] = (_Float16)(oa2[r] * inv);
    orow[48] = (_Float16)(oa3[r] * inv);
  }
}

// ---------------------------------------------------------------------------
// K5: output projection + residual: out[b,o,l] = x + sum_c Wp[o,c]*ot[l,c] + pb[o]
// ---------------------------------------------------------------------------
__global__ __launch_bounds__(32) void k_proj(const _Float16* __restrict__ Wp,
                                             const _Float16* __restrict__ Ot,
                                             const float* __restrict__ pb,
                                             const float* __restrict__ x,
                                             float* __restrict__ out) {
  __shared__ alignas(32) _Float16 Ts[2][64 * 32];
  const int o0 = blockIdx.x * 16;
  const int l0 = blockIdx.y * 64;
  const int b  = blockIdx.z;
  const int lane = threadIdx.x;
  const _Float16* Ap = Wp + o0 * C;
  const _Float16* Bp = Ot + ((size_t)b * L + l0) * C;
  v8f acc0 = {}, acc1 = {}, acc2 = {}, acc3 = {};

  tdm_load_2d(Ts[0], Bp, 32, 64, C);
  for (int kk = 0; kk < C; kk += 32) {
    const int cur = (kk >> 5) & 1;
    if (kk + 32 < C) {
      tdm_load_2d(Ts[cur ^ 1], Bp + kk + 32, 32, 64, C);
      __builtin_amdgcn_s_wait_tensorcnt(1);
    } else {
      __builtin_amdgcn_s_wait_tensorcnt(0);
    }
    __syncthreads();
    const v16h a = load_A_rowmajor(Ap + kk, C);
    const _Float16* T = Ts[cur];
    acc0 = wmma_f16(a, load_B_nmajor(T, 0),  acc0);
    acc1 = wmma_f16(a, load_B_nmajor(T, 16), acc1);
    acc2 = wmma_f16(a, load_B_nmajor(T, 32), acc2);
    acc3 = wmma_f16(a, load_B_nmajor(T, 48), acc3);
    __syncthreads();
  }
  const int n = lane & 15;
  const int mb = (lane >> 4) << 3;
  v8f accs[4] = {acc0, acc1, acc2, acc3};
#pragma unroll
  for (int j = 0; j < 4; ++j) {
#pragma unroll
    for (int r = 0; r < 8; ++r) {
      const int o = o0 + mb + r;
      const size_t idx = ((size_t)b * C + o) * L + l0 + j * 16 + n;
      out[idx] = x[idx] + accs[j][r] + pb[o];
    }
  }
}

// ---------------------------------------------------------------------------
// Launcher
// ---------------------------------------------------------------------------
extern "C" void kernel_launch(void* const* d_in, const int* in_sizes, int n_in,
                              void* d_out, int out_size, void* d_ws, size_t ws_size,
                              hipStream_t stream) {
  const float* x      = (const float*)d_in[0];
  const float* gn_w   = (const float*)d_in[1];
  const float* gn_b   = (const float*)d_in[2];
  const float* qkv_w  = (const float*)d_in[3];
  const float* qkv_b  = (const float*)d_in[4];
  const float* proj_w = (const float*)d_in[5];
  const float* proj_b = (const float*)d_in[6];
  float* out = (float*)d_out;

  char* w = (char*)d_ws;
  size_t off = 0;
  auto take = [&](size_t bytes) {
    char* p = w + off;
    off = (off + bytes + 255) & ~(size_t)255;
    return p;
  };
  float*     stats = (float*)take(BSZ * G * 2 * sizeof(float));
  _Float16*  wq    = (_Float16*)take((size_t)3 * C * C * sizeof(_Float16));
  _Float16*  wp    = (_Float16*)take((size_t)C * C * sizeof(_Float16));
  _Float16*  ht    = (_Float16*)take((size_t)BSZ * C * L * sizeof(_Float16));
  _Float16*  qb    = (_Float16*)take((size_t)BSZ * C * L * sizeof(_Float16));
  _Float16*  kb    = (_Float16*)take((size_t)BSZ * C * L * sizeof(_Float16));
  _Float16*  vt    = (_Float16*)take((size_t)BSZ * C * L * sizeof(_Float16));
  _Float16*  ot    = (_Float16*)take((size_t)BSZ * C * L * sizeof(_Float16));
  (void)ws_size; (void)in_sizes; (void)n_in; (void)out_size;

  k_convert_w<<<dim3((3 * C * C + 255) / 256), 256, 0, stream>>>(qkv_w, proj_w, wq, wp);
  k_gn_stats<<<dim3(BSZ * G), 256, 0, stream>>>(x, stats);
  k_gn_apply_t<<<dim3(L / 16, BSZ), 256, 0, stream>>>(x, stats, gn_w, gn_b, ht);
  k_qkv_gemm<<<dim3(3 * C / 16, L / 64, BSZ), 32, 0, stream>>>(wq, ht, qkv_b, qb, kb, vt);
  k_attn<<<dim3(L / 16, BSZ * NH), 32, 0, stream>>>(qb, kb, vt, ot);
  k_proj<<<dim3(C / 16, L / 64, BSZ), 32, 0, stream>>>(wp, ot, proj_b, x, out);
}
